// MultiHeadAttention_68736656605655
// MI455X (gfx1250) — matrix operand order repaired and checked by reading, not compiled, not run on hardware
//
#include <hip/hip_runtime.h>

// ---------------------------------------------------------------------------
// MHA forward for gfx1250 (MI455X): bf16 WMMA everywhere, TDM tile staging,
// flash attention with online softmax.
// ---------------------------------------------------------------------------

typedef __bf16 bf16;
typedef bf16  v16bf __attribute__((ext_vector_type(16)));
typedef bf16  v8bf  __attribute__((ext_vector_type(8)));
typedef bf16  v4bf  __attribute__((ext_vector_type(4)));
typedef float v8f   __attribute__((ext_vector_type(8)));
typedef unsigned int u32x4 __attribute__((ext_vector_type(4)));
typedef int   i32x8 __attribute__((ext_vector_type(8)));

#define BDIM  1024            // D
#define NHEAD 16
#define DHEAD 64
#define SEQ   2048
#define BATCH 2
#define MTOT  (BATCH * SEQ)   // 4096 rows

// concat two 8-wide bf16 vectors into a WMMA 16-wide operand
__device__ __forceinline__ v16bf cat8(v8bf lo, v8bf hi) {
  return __builtin_shufflevector(lo, hi, 0, 1, 2, 3, 4, 5, 6, 7,
                                 8, 9, 10, 11, 12, 13, 14, 15);
}

__device__ __forceinline__ v8f wmma_bf16(v16bf a, v16bf b, v8f c) {
  // (neg_a, A, neg_b, B, c_mod, C, reuse_a, reuse_b)
  return __builtin_amdgcn_wmma_f32_16x16x32_bf16(false, a, false, b,
                                                 (short)0, c, false, false);
}

// LDS byte offset of a __shared__ object (flat LDS aperture: offset in [31:0])
__device__ __forceinline__ unsigned int lds_off(const void* p) {
  return (unsigned int)(unsigned long long)p;
}

// ---------------------------------------------------------------------------
// TDM: DMA a [tile_rows x 32] bf16 tile (row stride K elems) from global into
// LDS with hardware padding: 16 DWORDs data + 4 DWORDs pad per row
// => LDS row stride 40 elems (80B), bank-conflict-free, 16B aligned.
// D# per CDNA5 ISA 8.3/8.4; groups 2/3 NULL (2D tensor).
// ---------------------------------------------------------------------------
__device__ __forceinline__ void tdm_load_tile_32w(
    unsigned int lds_addr, const bf16* gptr, int tile_rows, int rows, int K) {
  unsigned long long ga = (unsigned long long)gptr;
  u32x4 g0;
  g0[0] = 1u;                                   // count=1, user descriptor
  g0[1] = lds_addr;                             // lds_addr (bytes)
  g0[2] = (unsigned int)ga;                     // global_addr[31:0]
  g0[3] = (unsigned int)((ga >> 32) & 0x01FFFFFFu) | (2u << 30);  // [56:32],type=2
  i32x8 g1;
  g1[0] = (1 << 16)            // data_size = 2 bytes
        | (1 << 20)            // pad_enable
        | (3 << 22)            // pad_interval: 16 DWORDs (one 32-elem row)
        | (3 << 25);           // pad_amount: 4 DWORDs (8 elems)
  g1[1] = (int)(((unsigned)K & 0xFFFFu) << 16);            // tensor_dim0[15:0]
  g1[2] = (int)((((unsigned)K >> 16) & 0xFFFFu)            // tensor_dim0[31:16]
        | (((unsigned)rows & 0xFFFFu) << 16));             // tensor_dim1[15:0]
  g1[3] = (int)((((unsigned)rows >> 16) & 0xFFFFu)         // tensor_dim1[31:16]
        | (32u << 16));                                    // tile_dim0 = 32
  g1[4] = tile_rows;                                       // tile_dim1
  g1[5] = K;                                               // dim0_stride[31:0]
  g1[6] = 0;
  g1[7] = 0;
  asm volatile("tensor_load_to_lds %0, %1" :: "s"(g0), "s"(g1) : "memory");
}

// ---------------------------------------------------------------------------
// fp32 -> bf16 cast (n divisible by 4)
// ---------------------------------------------------------------------------
__global__ __launch_bounds__(256) void cast_f32_to_bf16(
    const float* __restrict__ x, bf16* __restrict__ y, int n) {
  int i = (blockIdx.x * blockDim.x + threadIdx.x) * 4;
  if (i >= n) return;
  float4 v = *(const float4*)(x + i);
  v4bf o;
  o[0] = (bf16)v.x; o[1] = (bf16)v.y; o[2] = (bf16)v.z; o[3] = (bf16)v.w;
  *(v4bf*)(y + i) = o;
}

// ---------------------------------------------------------------------------
// C[M,N] = A[M,K] @ W[N,K]^T   (bf16 operands, fp32 accum)
// 256 threads = 8 waves (2x4); block tile 128x256; wave tile 64x64
// (4x4 fragments => 16 WMMAs per K-step, 2.0 WMMA per fragment load).
// K step 32. Tiles staged by TDM (wave 0 issues, TENSORcnt-synced), LDS
// double-buffered (60KB); one barrier per K step.
// ---------------------------------------------------------------------------
#define GBK  32
#define LDST 40   // LDS row stride in elems (32 data + 8 pad from TDM)

__global__ __launch_bounds__(256) void gemm_bf16_xwt(
    const bf16* __restrict__ A, const bf16* __restrict__ W,
    bf16* __restrict__ Cb, float* __restrict__ Cf,
    int M, int N, int K) {
  __shared__ bf16 lA[2][128 * LDST];
  __shared__ bf16 lB[2][256 * LDST];

  const int tid  = threadIdx.x;
  const int lane = tid & 31;
  const int wid  = tid >> 5;
  const int l15  = lane & 15;
  const int oa   = (lane < 16) ? 0 : 8;    // A-frag K-chunk select
  const int ob   = (lane < 16) ? 0 : 8;    // B-frag K-chunk select
  const int m0   = blockIdx.y * 128;
  const int n0   = blockIdx.x * 256;
  const int mw   = (wid >> 2) * 64;
  const int nw   = (wid & 3) * 64;

  v8f acc[4][4];
#pragma unroll
  for (int i = 0; i < 4; ++i)
#pragma unroll
    for (int j = 0; j < 4; ++j)
#pragma unroll
      for (int r = 0; r < 8; ++r) acc[i][j][r] = 0.f;

  if (wid == 0) {  // prime buffer 0
    tdm_load_tile_32w(lds_off(lA[0]), A + (size_t)m0 * K, 128, M, K);
    tdm_load_tile_32w(lds_off(lB[0]), W + (size_t)n0 * K, 256, N, K);
  }

  for (int k0 = 0; k0 < K; k0 += GBK) {
    const int cur = (k0 >> 5) & 1;
    __builtin_amdgcn_s_wait_tensorcnt(0);   // wave0: DMA of buf[cur] done
    __syncthreads();                        // everyone: buf[cur] visible,
                                            // prior reads of buf[cur^1] done
    if (wid == 0 && k0 + GBK < K) {         // kick DMA of next tile
      tdm_load_tile_32w(lds_off(lA[cur ^ 1]),
                        A + (size_t)m0 * K + k0 + GBK, 128, M, K);
      tdm_load_tile_32w(lds_off(lB[cur ^ 1]),
                        W + (size_t)n0 * K + k0 + GBK, 256, N, K);
    }

    const bf16* la = lA[cur];
    const bf16* lb = lB[cur];
    v16bf bfr[4], afr[4];
#pragma unroll
    for (int j = 0; j < 4; ++j) {
      const bf16* p = &lb[(nw + 16 * j + l15) * LDST + ob];
      bfr[j] = cat8(*(const v8bf*)p, *(const v8bf*)(p + 16));
    }
#pragma unroll
    for (int i = 0; i < 4; ++i) {
      const bf16* p = &la[(mw + 16 * i + l15) * LDST + oa];
      afr[i] = cat8(*(const v8bf*)p, *(const v8bf*)(p + 16));
    }
#pragma unroll
    for (int i = 0; i < 4; ++i)
#pragma unroll
      for (int j = 0; j < 4; ++j)
        acc[i][j] = wmma_bf16(afr[i], bfr[j], acc[i][j]);
  }

  const int rbase = (lane < 16) ? 0 : 8;
#pragma unroll
  for (int i = 0; i < 4; ++i)
#pragma unroll
    for (int j = 0; j < 4; ++j)
#pragma unroll
      for (int r = 0; r < 8; ++r) {
        size_t row = (size_t)(m0 + mw + 16 * i + r + rbase);
        size_t col = (size_t)(n0 + nw + 16 * j + l15);
        if (Cf) Cf[row * N + col] = acc[i][j][r];
        else    Cb[row * N + col] = (bf16)acc[i][j][r];
      }
}

// ---------------------------------------------------------------------------
// Flash attention, causal, one head per blockIdx.y, 64 q-rows per block
// (4 waves x 16 rows). Key blocks of 32. Q/K fragments load straight from
// global (contiguous along dh); V is transposed into shared LDS; P round-trips
// through per-wave LDS to become an A-fragment. fp32 online softmax.
// ---------------------------------------------------------------------------
#define VTL 40

__global__ __launch_bounds__(128) void flash_attn_kernel(
    const bf16* __restrict__ Q, const bf16* __restrict__ K,
    const bf16* __restrict__ V, bf16* __restrict__ O) {
  __shared__ bf16 VT[DHEAD * VTL];        // V^T tile: [dh][key]
  __shared__ bf16 PT[4][16 * VTL];        // per-wave P tile: [qrow][key]

  const int tid  = threadIdx.x;
  const int lane = tid & 31;
  const int wid  = tid >> 5;
  const int l15  = lane & 15;
  const int hi   = (lane < 16) ? 0 : 1;
  const int oa   = hi ? 8 : 0;
  const int ob   = hi ? 8 : 0;

  const int b = blockIdx.y / NHEAD;
  const int h = blockIdx.y % NHEAD;
  const int q0blk = blockIdx.x * 64;
  const int q0w   = q0blk + wid * 16;

  const size_t base = (size_t)b * SEQ * BDIM + (size_t)h * DHEAD;

  // Q A-fragments for the two dh steps (dh 0..31, 32..63)
  const bf16* qrow = Q + base + (size_t)(q0w + l15) * BDIM;
  v16bf qf[2];
#pragma unroll
  for (int s = 0; s < 2; ++s) {
    const bf16* p = qrow + 32 * s + oa;
    qf[s] = cat8(*(const v8bf*)p, *(const v8bf*)(p + 16));
  }

  v8f o[4];
#pragma unroll
  for (int f = 0; f < 4; ++f)
#pragma unroll
    for (int r = 0; r < 8; ++r) o[f][r] = 0.f;
  float mrow[8], lrow[8];
#pragma unroll
  for (int r = 0; r < 8; ++r) { mrow[r] = -1e30f; lrow[r] = 0.f; }

  const int kend = q0blk + 64;  // causal: keys strictly beyond block unused
  for (int kb = 0; kb < kend; kb += 32) {
    // cooperative V^T staging (coalesced global read, transposed LDS write)
    for (int idx = tid; idx < 32 * DHEAD; idx += 128) {
      int key = idx >> 6;
      int dh  = idx & 63;
      VT[dh * VTL + key] = V[base + (size_t)(kb + key) * BDIM + dh];
    }
    __syncthreads();

    if (kb <= q0w + 15) {  // wave-uniform: EXEC stays all-ones for WMMA
      // scores = Q K^T  (2 key sub-tiles x 2 dh steps)
      v8f sc[2];
#pragma unroll
      for (int j = 0; j < 2; ++j) {
#pragma unroll
        for (int r = 0; r < 8; ++r) sc[j][r] = 0.f;
        const bf16* krow = K + base + (size_t)(kb + 16 * j + l15) * BDIM;
#pragma unroll
        for (int s = 0; s < 2; ++s) {
          const bf16* p = krow + 32 * s + ob;
          v16bf kf = cat8(*(const v8bf*)p, *(const v8bf*)(p + 16));
          sc[j] = wmma_bf16(qf[s], kf, sc[j]);
        }
      }

      // causal mask + online softmax (rows live in 16-lane halves)
#pragma unroll
      for (int r = 0; r < 8; ++r) {
        const int qi = q0w + r + (hi ? 8 : 0);
        float s0 = sc[0][r] * 0.125f;              // 1/sqrt(64)
        float s1 = sc[1][r] * 0.125f;
        if (kb + l15 > qi)      s0 = -1e30f;
        if (kb + 16 + l15 > qi) s1 = -1e30f;
        float rm = fmaxf(s0, s1);
#pragma unroll
        for (int d = 1; d < 16; d <<= 1) rm = fmaxf(rm, __shfl_xor(rm, d, 16));
        float mn = fmaxf(mrow[r], rm);
        float corr = __expf(mrow[r] - mn);
        float p0 = __expf(s0 - mn);
        float p1 = __expf(s1 - mn);
        float rs = p0 + p1;
#pragma unroll
        for (int d = 1; d < 16; d <<= 1) rs += __shfl_xor(rs, d, 16);
        lrow[r] = lrow[r] * corr + rs;
        mrow[r] = mn;
#pragma unroll
        for (int f = 0; f < 4; ++f) o[f][r] *= corr;
        const int prow = r + (hi ? 8 : 0);
        PT[wid][prow * VTL + l15]      = (bf16)p0;
        PT[wid][prow * VTL + 16 + l15] = (bf16)p1;
      }

      // P (A-frag from per-wave LDS, same-wave DS ordering) x V^T
      const bf16* pr = &PT[wid][l15 * VTL + oa];
      v16bf pf = cat8(*(const v8bf*)pr, *(const v8bf*)(pr + 16));
#pragma unroll
      for (int f = 0; f < 4; ++f) {
        const bf16* vr = &VT[(16 * f + l15) * VTL + ob];
        v16bf vf = cat8(*(const v8bf*)vr, *(const v8bf*)(vr + 16));
        o[f] = wmma_bf16(pf, vf, o[f]);
      }
    }
    __syncthreads();
  }

  // normalize and write in (B, T, H*DH) layout
#pragma unroll
  for (int f = 0; f < 4; ++f)
#pragma unroll
    for (int r = 0; r < 8; ++r) {
      const int row = q0w + r + (hi ? 8 : 0);
      float val = o[f][r] / lrow[r];
      O[base + (size_t)row * BDIM + 16 * f + l15] = (bf16)val;
    }
}

// ---------------------------------------------------------------------------
extern "C" void kernel_launch(void* const* d_in, const int* in_sizes, int n_in,
                              void* d_out, int out_size, void* d_ws, size_t ws_size,
                              hipStream_t stream) {
  const float* q  = (const float*)d_in[0];
  const float* k  = (const float*)d_in[1];
  const float* v  = (const float*)d_in[2];
  const float* wq = (const float*)d_in[3];
  const float* wk = (const float*)d_in[4];
  const float* wv = (const float*)d_in[5];
  const float* wo = (const float*)d_in[6];

  const size_t NTOK = (size_t)MTOT * BDIM;  // 4 Mi elems
  const size_t NW   = (size_t)BDIM * BDIM;  // 1 Mi elems

  bf16* p   = (bf16*)d_ws;
  bf16* qb  = p; p += NTOK;
  bf16* kb  = p; p += NTOK;
  bf16* vb  = p; p += NTOK;
  bf16* wqb = p; p += NW;
  bf16* wkb = p; p += NW;
  bf16* wvb = p; p += NW;
  bf16* wob = p; p += NW;
  bf16* Qp  = p; p += NTOK;
  bf16* Kp  = p; p += NTOK;
  bf16* Vp  = p; p += NTOK;
  bf16* AO  = qb;  // reuse: qb dead after Q projection

  auto cast = [&](const float* x, bf16* y, size_t n) {
    cast_f32_to_bf16<<<dim3((unsigned)((n / 4 + 255) / 256)), dim3(256), 0,
                       stream>>>(x, y, (int)n);
  };
  cast(q, qb, NTOK);  cast(k, kb, NTOK);  cast(v, vb, NTOK);
  cast(wq, wqb, NW);  cast(wk, wkb, NW);  cast(wv, wvb, NW); cast(wo, wob, NW);

  dim3 ggrid(BDIM / 256, MTOT / 128);
  gemm_bf16_xwt<<<ggrid, dim3(256), 0, stream>>>(qb, wqb, Qp, nullptr,
                                                 MTOT, BDIM, BDIM);
  gemm_bf16_xwt<<<ggrid, dim3(256), 0, stream>>>(kb, wkb, Kp, nullptr,
                                                 MTOT, BDIM, BDIM);
  gemm_bf16_xwt<<<ggrid, dim3(256), 0, stream>>>(vb, wvb, Vp, nullptr,
                                                 MTOT, BDIM, BDIM);

  flash_attn_kernel<<<dim3(SEQ / 64, BATCH * NHEAD), dim3(128), 0, stream>>>(
      Qp, Kp, Vp, AO);

  gemm_bf16_xwt<<<ggrid, dim3(256), 0, stream>>>(AO, wob, nullptr,
                                                 (float*)d_out, MTOT, BDIM, BDIM);
}
